// MuSelectorModel_53506702573827
// MI455X (gfx1250) — matrix operand order repaired and checked by reading, not compiled, not run on hardware
//
#include <hip/hip_runtime.h>
#include <hip/hip_bf16.h>
#include <math.h>

// ---------------- model constants (from reference) ----------------
#define BATCH 32
#define SEQL  2048
#define HID   68
#define DIN   136          // 2*HID
#define D2    272          // 2*DIN (xz width)
#define DST   16
#define DCV   4
#define DTR   5
#define XD    37           // DTR + 2*DST
#define NOUT  128
#define MROWS (BATCH * SEQL)   // 65536, multiple of 64

typedef __attribute__((ext_vector_type(16))) _Float16 v16h;
typedef __attribute__((ext_vector_type(8)))  _Float16 v8h;
typedef __attribute__((ext_vector_type(8)))  float    v8f;

__device__ __forceinline__ float fast_rcp(float x)   { return __builtin_amdgcn_rcpf(x); }
__device__ __forceinline__ float fsigmoid(float x)   { return fast_rcp(1.f + __expf(-x)); }
__device__ __forceinline__ float fsilu(float x)      { return x * fsigmoid(x); }

static __host__ __device__ inline int roundup(int v, int a) { return (v + a - 1) / a * a; }

// =====================================================================
// Pack A: f32 [M,K] -> f16 [M,K32] with optional fused sigmoid, zero-pad
// K..K32. Branch-free: clamped load + select.
// =====================================================================
template <int ACT_SIGMOID>
__global__ void pack_a_kernel(const float* __restrict__ A, _Float16* __restrict__ Ah,
                              int K, int K32, size_t total /* M*K32 */)
{
    size_t idx = (size_t)blockIdx.x * blockDim.x + threadIdx.x;
    if (idx >= total) return;
    int    k = (int)(idx % K32);
    size_t m = idx / K32;
    int    kc = (k < K) ? k : (K - 1);
    float  v = A[m * K + kc];
    if (ACT_SIGMOID) v = fsigmoid(v);
    v = (k < K) ? v : 0.f;
    Ah[idx] = (_Float16)v;
}

// =====================================================================
// Pack W: f32 [N,K] -> f16 [Npad,K32], zero-padded in both dims.
// =====================================================================
__global__ void pack_w_kernel(const float* __restrict__ W, _Float16* __restrict__ Wh,
                              int N, int K, int K32, size_t total /* Npad*K32 */)
{
    size_t idx = (size_t)blockIdx.x * blockDim.x + threadIdx.x;
    if (idx >= total) return;
    int    k = (int)(idx % K32);
    int    n = (int)(idx / K32);
    int    kc = (k < K) ? k : (K - 1);
    int    nc = (n < N) ? n : (N - 1);
    float  v = W[(size_t)nc * K + kc];
    v = (k < K && n < N) ? v : 0.f;
    Wh[idx] = (_Float16)v;
}

// =====================================================================
// Guard-free WMMA GEMM on packed f16:
//   C[M,N] = Ah[M,K32] @ Wh[Npad,K32]^T (+ bias[N])
// One wave32 per 16x16 tile; 4 waves (4 M-tiles) per block.
//
// Fragment layouts used here (A and B share contiguous k = hi*16 + s):
//  A 16x32 f16: lane (m = lane&15, hi = lane>>4):
//      halves 0..7  = K[k0 + hi*16     .. +7]   (contiguous, 16B)
//      halves 8..15 = K[k0 + hi*16 + 8 .. +7]   (contiguous, 16B)
//  B 32x16 f16: lane (n = lane&15, hi = lane>>4):
//      halves 0..15 = K[k0 + hi*16 .. +15]      (contiguous, 32B)
//  C/D f32 16x16: lane (n = lane&15), VGPR v -> M = v + 8*(lane>>4).
// =====================================================================
__global__ void gemm_wmma_packed(const _Float16* __restrict__ Ah,
                                 const _Float16* __restrict__ Wh,
                                 const float* __restrict__ bias,
                                 float* __restrict__ C,
                                 int N, int K32)
{
    const int lane  = threadIdx.x;            // 0..31
    const int half  = lane >> 4;              // 0 | 1
    const int l15   = lane & 15;
    const int mBase = (blockIdx.y * 4 + threadIdx.y) * 16;
    const int nRow  = blockIdx.x * 16 + l15;

    const _Float16* pa = Ah + (size_t)(mBase + l15) * K32;
    const _Float16* pw = Wh + (size_t)nRow * K32 + half * 16;

    v8f acc = {};
    for (int k0 = 0; k0 < K32; k0 += 32) {
        v8h a0 = *(const v8h*)(pa + k0 + half * 16);       // b128
        v8h a1 = *(const v8h*)(pa + k0 + half * 16 + 8);   // b128
        v16h a, b;
        #pragma unroll
        for (int i = 0; i < 8; ++i) { a[i] = a0[i]; a[8 + i] = a1[i]; }
        b = *(const v16h*)(pw + k0);                       // 2x b128
        acc = __builtin_amdgcn_wmma_f32_16x16x32_f16(
                  false, a, false, b, (short)0, acc, false, false);
    }

    if (nRow < N) {
        const float bv = bias ? bias[nRow] : 0.f;
        #pragma unroll
        for (int v = 0; v < 8; ++v)
            C[(size_t)(mBase + v + 8 * half) * N + nRow] = acc[v] + bv;
    }
}

// =====================================================================
// Embedding: h[b,t,:] = [ipt0/127, ipt1*(L/12), 0] @ W0^T + b0   (K=3)
// =====================================================================
__global__ void embed_kernel(const float* __restrict__ ipt,
                             const float* __restrict__ W0,
                             const float* __restrict__ b0,
                             float* __restrict__ h)
{
    size_t idx = (size_t)blockIdx.x * blockDim.x + threadIdx.x;
    size_t total = (size_t)MROWS * HID;
    if (idx >= total) return;
    int    nn = (int)(idx % HID);
    size_t bt = idx / HID;
    float c0 = ipt[bt * 3 + 0] * (1.0f / 127.0f);
    float c1 = ipt[bt * 3 + 1] * ((float)SEQL / 12.0f);   // /tnf == *(L/12)
    h[idx] = c0 * W0[nn * 3 + 0] + c1 * W0[nn * 3 + 1] + b0[nn];
}

// =====================================================================
// Causal depthwise conv (width 4) + SiLU.  xi lives in xz[..., 0:DIN].
// =====================================================================
__global__ void conv_silu_kernel(const float* __restrict__ xz,
                                 const float* __restrict__ cw,
                                 const float* __restrict__ cb,
                                 float* __restrict__ out)
{
    size_t idx = (size_t)blockIdx.x * blockDim.x + threadIdx.x;
    size_t total = (size_t)MROWS * DIN;
    if (idx >= total) return;
    int    d  = (int)(idx % DIN);
    size_t bt = idx / DIN;
    int    t  = (int)(bt % SEQL);
    size_t b  = bt / SEQL;
    const float* base = xz + b * (size_t)SEQL * D2;
    float acc = cb[d];
    #pragma unroll
    for (int j = 0; j < DCV; ++j) {
        int tt = t - (DCV - 1) + j;
        if (tt >= 0) acc += base[(size_t)tt * D2 + d] * cw[d * DCV + j];
    }
    out[idx] = fsilu(acc);
}

// =====================================================================
// delta = softplus(dt @ dt_w^T + dt_b)   (K = DTR = 5, tiny -> VALU)
// =====================================================================
__global__ void delta_kernel(const float* __restrict__ xdbl,
                             const float* __restrict__ dtw,
                             const float* __restrict__ dtb,
                             float* __restrict__ out)
{
    size_t idx = (size_t)blockIdx.x * blockDim.x + threadIdx.x;
    size_t total = (size_t)MROWS * DIN;
    if (idx >= total) return;
    int    d  = (int)(idx % DIN);
    size_t bt = idx / DIN;
    const float* s = xdbl + bt * XD;
    float acc = dtb[d];
    #pragma unroll
    for (int r = 0; r < DTR; ++r) acc += s[r] * dtw[d * DTR + r];
    out[idx] = (acc > 20.f) ? acc : log1pf(__expf(acc));   // softplus
}

// =====================================================================
// Selective scan. grid = (DIN/8, B); block = 128 (= 8 channels x 16 states).
// Lanes within a 16-lane half reduce over DST with __shfl_xor (wave32;
// masks 1/2/4/8 never cross the 16-lane channel boundary).
// =====================================================================
__global__ void scan_kernel(const float* __restrict__ delta,   // [B,L,DIN]
                            const float* __restrict__ xconv,   // [B,L,DIN]
                            const float* __restrict__ xdbl,    // [B,L,XD]
                            const float* __restrict__ A_log,   // [DIN,DST]
                            float* __restrict__ ys)            // [B,L,DIN]
{
    const int n  = threadIdx.x & 15;          // state index
    const int dl = threadIdx.x >> 4;          // 0..7 channel-in-block
    const int d  = blockIdx.x * 8 + dl;       // channel 0..135
    const int b  = blockIdx.y;

    const float Ad = -__expf(A_log[d * DST + n]);
    float h = 0.f;

    const float* dptr = delta + (size_t)b * SEQL * DIN;
    const float* xptr = xconv + (size_t)b * SEQL * DIN;
    const float* sptr = xdbl  + (size_t)b * SEQL * XD;
    float*       yptr = ys    + (size_t)b * SEQL * DIN;

    for (int t = 0; t < SEQL; ++t) {
        float dv = dptr[(size_t)t * DIN + d];
        float xv = xptr[(size_t)t * DIN + d];
        float Bn = sptr[(size_t)t * XD + DTR + n];
        float Cn = sptr[(size_t)t * XD + DTR + DST + n];
        float dA = __expf(dv * Ad);
        h = dA * h + dv * xv * Bn;
        float p = h * Cn;
        p += __shfl_xor(p, 1, 32);
        p += __shfl_xor(p, 2, 32);
        p += __shfl_xor(p, 4, 32);
        p += __shfl_xor(p, 8, 32);
        if (n == 0) yptr[(size_t)t * DIN + d] = p;
    }
}

// =====================================================================
// y = (ys + xconv * D) * silu(z)   with z = xz[..., DIN:2*DIN]
// =====================================================================
__global__ void gate_kernel(const float* __restrict__ ys,
                            const float* __restrict__ xconv,
                            const float* __restrict__ xz,
                            const float* __restrict__ Dp,
                            float* __restrict__ y)
{
    size_t idx = (size_t)blockIdx.x * blockDim.x + threadIdx.x;
    size_t total = (size_t)MROWS * DIN;
    if (idx >= total) return;
    int    d  = (int)(idx % DIN);
    size_t bt = idx / DIN;
    float z = xz[bt * D2 + DIN + d];
    y[idx] = (ys[idx] + xconv[idx] * Dp[d]) * fsilu(z);
}

// =====================================================================
// Head: logits = h[:, L-1, :] @ W1^T + b1, softmax. Block per batch row.
// =====================================================================
__global__ void head_kernel(const float* __restrict__ h,   // [B,L,HID]
                            const float* __restrict__ W1,  // [NOUT,HID]
                            const float* __restrict__ b1,
                            float* __restrict__ out)       // [B,NOUT]
{
    __shared__ float red[NOUT];
    const int b  = blockIdx.x;
    const int nn = threadIdx.x;
    const float* hv = h + ((size_t)b * SEQL + (SEQL - 1)) * HID;
    float acc = b1[nn];
    for (int k = 0; k < HID; ++k) acc += hv[k] * W1[nn * HID + k];
    red[nn] = acc;
    __syncthreads();
    for (int s = NOUT / 2; s > 0; s >>= 1) {
        if (nn < s) red[nn] = fmaxf(red[nn], red[nn + s]);
        __syncthreads();
    }
    float mx = red[0];
    __syncthreads();
    float e = __expf(acc - mx);
    red[nn] = e;
    __syncthreads();
    for (int s = NOUT / 2; s > 0; s >>= 1) {
        if (nn < s) red[nn] += red[nn + s];
        __syncthreads();
    }
    out[(size_t)b * NOUT + nn] = e * fast_rcp(red[0]);
}

// =====================================================================
// Launcher
// =====================================================================
struct GemmBufs { _Float16* apack; _Float16* wpack; };

static void run_gemm(const float* A, const float* W, const float* bias, float* C,
                     int N, int K, int actSigmoid, GemmBufs bufs, hipStream_t stream)
{
    const int K32  = roundup(K, 32);
    const int Npad = roundup(N, 16);
    const int TB   = 256;

    size_t nA = (size_t)MROWS * K32;
    if (actSigmoid)
        pack_a_kernel<1><<<dim3((nA + TB - 1) / TB), dim3(TB), 0, stream>>>(A, bufs.apack, K, K32, nA);
    else
        pack_a_kernel<0><<<dim3((nA + TB - 1) / TB), dim3(TB), 0, stream>>>(A, bufs.apack, K, K32, nA);

    size_t nW = (size_t)Npad * K32;
    pack_w_kernel<<<dim3((unsigned)((nW + TB - 1) / TB)), dim3(TB), 0, stream>>>(W, bufs.wpack, N, K, K32, nW);

    dim3 grid(Npad / 16, MROWS / 64);
    gemm_wmma_packed<<<grid, dim3(32, 4), 0, stream>>>(bufs.apack, bufs.wpack, bias, C, N, K32);
}

extern "C" void kernel_launch(void* const* d_in, const int* in_sizes, int n_in,
                              void* d_out, int out_size, void* d_ws, size_t ws_size,
                              hipStream_t stream)
{
    (void)in_sizes; (void)n_in; (void)out_size; (void)ws_size;

    const float* ipt = (const float*)d_in[0];
    const float* W0  = (const float*)d_in[1];
    const float* b0  = (const float*)d_in[2];
    const float* W1  = (const float*)d_in[25];
    const float* b1  = (const float*)d_in[26];
    float* out = (float*)d_out;

    // ---- workspace carve-out, 64B aligned ----
    char*  base = (char*)d_ws;
    size_t off  = 0;
    auto carve = [&](size_t bytes) -> char* {
        char* p = base + off;
        off += (bytes + 63) & ~(size_t)63;
        return p;
    };
    float* hbuf = (float*)carve((size_t)MROWS * HID * 4);
    float* xz   = (float*)carve((size_t)MROWS * D2  * 4);
    float* xcv  = (float*)carve((size_t)MROWS * DIN * 4);
    float* xdb  = (float*)carve((size_t)MROWS * XD  * 4);
    float* dlt  = (float*)carve((size_t)MROWS * DIN * 4);
    float* ysb  = (float*)carve((size_t)MROWS * DIN * 4);
    float* yg   = (float*)carve((size_t)MROWS * DIN * 4);
    float* htmp = (float*)carve((size_t)MROWS * HID * 4);
    GemmBufs gb;
    gb.apack = (_Float16*)carve((size_t)MROWS * 160 * 2);  // max K32 = 160
    gb.wpack = (_Float16*)carve((size_t)272 * 160 * 2);    // max Npad*K32

    const int TB = 256;
    const size_t nHID = (size_t)MROWS * HID;
    const size_t nDIN = (size_t)MROWS * DIN;

    // ---- embed ----
    embed_kernel<<<dim3((nHID + TB - 1) / TB), dim3(TB), 0, stream>>>(ipt, W0, b0, hbuf);

    // ---- two mamba blocks ----
    for (int blk = 0; blk < 2; ++blk) {
        const int bi = 3 + blk * 11;
        const float* in_w   = (const float*)d_in[bi + 0];  // [D2, HID]
        const float* conv_w = (const float*)d_in[bi + 1];  // [DIN, DCV]
        const float* conv_b = (const float*)d_in[bi + 2];  // [DIN]
        const float* x_w    = (const float*)d_in[bi + 3];  // [XD, DIN]
        const float* dt_w   = (const float*)d_in[bi + 4];  // [DIN, DTR]
        const float* dt_b   = (const float*)d_in[bi + 5];  // [DIN]
        const float* A_log  = (const float*)d_in[bi + 6];  // [DIN, DST]
        const float* Dp     = (const float*)d_in[bi + 7];  // [DIN]
        const float* out_w  = (const float*)d_in[bi + 8];  // [HID, DIN]
        const float* l_w    = (const float*)d_in[bi + 9];  // [HID, HID]
        const float* l_b    = (const float*)d_in[bi + 10]; // [HID]

        // xz = sigmoid(h) @ in_w^T          [M, D2]
        run_gemm(hbuf, in_w, nullptr, xz, D2, HID, /*sigmoid=*/1, gb, stream);

        // causal depthwise conv + silu      [M, DIN]
        conv_silu_kernel<<<dim3((nDIN + TB - 1) / TB), dim3(TB), 0, stream>>>(xz, conv_w, conv_b, xcv);

        // x_dbl = xcv @ x_w^T               [M, XD]
        run_gemm(xcv, x_w, nullptr, xdb, XD, DIN, 0, gb, stream);

        // delta = softplus(dt @ dt_w^T + dt_b)
        delta_kernel<<<dim3((nDIN + TB - 1) / TB), dim3(TB), 0, stream>>>(xdb, dt_w, dt_b, dlt);

        // selective scan                     [M, DIN]
        scan_kernel<<<dim3(DIN / 8, BATCH), dim3(128), 0, stream>>>(dlt, xcv, xdb, A_log, ysb);

        // gate: y = (ys + xcv*D) * silu(z)
        gate_kernel<<<dim3((nDIN + TB - 1) / TB), dim3(TB), 0, stream>>>(ysb, xcv, xz, Dp, yg);

        // h' = y @ out_w^T                   [M, HID]
        run_gemm(yg, out_w, nullptr, htmp, HID, DIN, 0, gb, stream);

        // h = sigmoid(h') @ l_w^T + l_b      [M, HID]
        run_gemm(htmp, l_w, l_b, hbuf, HID, HID, 1, gb, stream);
    }

    // ---- head + softmax ----
    head_kernel<<<dim3(BATCH), dim3(NOUT), 0, stream>>>(hbuf, W1, b1, out);
}
